// FlareTGCN_61607010894575
// MI455X (gfx1250) — compile-verified, matrix operand order checked by reading, not run on hardware
//
#include <hip/hip_runtime.h>
#include <hip/hip_bf16.h>

// ---------------------------------------------------------------------------
// FlareTGCN for MI455X (gfx1250): WMMA-tiled GEMMs + atomic segment-sum graph
// aggregation + fused GRU elementwise kernels. All launches on `stream`.
// ---------------------------------------------------------------------------

typedef __attribute__((ext_vector_type(16))) _Float16 v16h;
typedef __attribute__((ext_vector_type(8)))  _Float16 v8h;
typedef __attribute__((ext_vector_type(8)))  float    v8f;

#define TM 64
#define TN 64
#define TK 32
#define LDSK 40   // padded K-stride (halfs): 80B rows -> bank stride 20, conflict-free frags

// C[M,Nout] (+)= A[M,K] @ W[K,Nout], fp32 in/out, f16 WMMA inner product.
// Block: 128 threads (4 waves) -> 64x64 C tile; wave w owns rows w*16..w*16+15.
// A staged row-major [m][k]; B staged TRANSPOSED [n][k] so both fragment reads
// are two contiguous ds_load_b128 per the ISA 16-bit 16x32 operand layout.
__global__ __launch_bounds__(128) void gemm_wmma_f16(
    const float* __restrict__ A, int lda,
    const float* __restrict__ W, int ldw,
    float* __restrict__ C, int ldc,
    int M, int K, int Nout, int accumulate)
{
    __shared__ _Float16 As[TM * LDSK];   // [64][40]
    __shared__ _Float16 Bs[TN * LDSK];   // [64][40]  (Bs[n][k] = W[k][n])

    const int t    = threadIdx.x;        // 0..127
    const int wave = t >> 5;             // 0..3
    const int lane = t & 31;
    const int m0   = blockIdx.x * TM;
    const int n0   = blockIdx.y * TN;

    v8f acc[4] = {{}, {}, {}, {}};

    for (int k0 = 0; k0 < K; k0 += TK) {
        // ---- stage A tile (64x32): thread -> row t>>1, 16 cols -------------
        {
            const int r    = t >> 1;
            const int cb   = (t & 1) * 16;
            const int grow = m0 + r;
            __align__(16) _Float16 tmp[16];
            if (grow < M) {
                const float* sp = A + (size_t)grow * lda + k0 + cb;
                if (k0 + TK < K) __builtin_prefetch(sp + TK, 0, 0);
#pragma unroll
                for (int i = 0; i < 16; i += 4) {
                    float4 v = *(const float4*)(sp + i);
                    tmp[i + 0] = (_Float16)v.x; tmp[i + 1] = (_Float16)v.y;
                    tmp[i + 2] = (_Float16)v.z; tmp[i + 3] = (_Float16)v.w;
                }
            } else {
#pragma unroll
                for (int i = 0; i < 16; ++i) tmp[i] = (_Float16)0.0f;
            }
            v8h* dp = (v8h*)(As + r * LDSK + cb);
            dp[0] = *(const v8h*)(tmp);
            dp[1] = *(const v8h*)(tmp + 8);
        }
        // ---- stage B tile transposed (Bs[n][k]): thread -> col n = t&63 ----
        {
            const int n  = t & 63;
            const int kh = (t >> 6) * 16;          // 0 or 16
            const float* sp = W + (size_t)(k0 + kh) * ldw + n0 + n;
            __align__(16) _Float16 tmp[16];
#pragma unroll
            for (int i = 0; i < 16; ++i)
                tmp[i] = (_Float16)sp[(size_t)i * ldw];
            v8h* dp = (v8h*)(Bs + n * LDSK + kh);
            dp[0] = *(const v8h*)(tmp);
            dp[1] = *(const v8h*)(tmp + 8);
        }
        __syncthreads();

        // ---- fragments per ISA 16-bit 16x32 layout (two b128 loads each) ---
        const int kb   = (lane < 16) ? 0 : 8;
        const int mloc = (lane & 15) + wave * 16;
        const _Float16* ap = As + mloc * LDSK + kb;
        v8h alo = *(const v8h*)(ap);
        v8h ahi = *(const v8h*)(ap + 16);
        v16h afrag = __builtin_shufflevector(alo, ahi,
            0, 1, 2, 3, 4, 5, 6, 7, 8, 9, 10, 11, 12, 13, 14, 15);

#pragma unroll
        for (int nt = 0; nt < 4; ++nt) {
            const int nloc = (lane & 15) + nt * 16;
            const _Float16* bp = Bs + nloc * LDSK + kb;
            v8h blo = *(const v8h*)(bp);
            v8h bhi = *(const v8h*)(bp + 16);
            v16h bfrag = __builtin_shufflevector(blo, bhi,
                0, 1, 2, 3, 4, 5, 6, 7, 8, 9, 10, 11, 12, 13, 14, 15);
            acc[nt] = __builtin_amdgcn_wmma_f32_16x16x32_f16(
                false, afrag, false, bfrag, (short)0, acc[nt], false, false);
        }
        __syncthreads();
    }

    // ---- store C per 8-VGPR f32 C/D layout --------------------------------
    const int rbase = m0 + wave * 16 + ((lane >= 16) ? 8 : 0);
    const int cl    = lane & 15;
#pragma unroll
    for (int nt = 0; nt < 4; ++nt) {
        const int col = n0 + nt * 16 + cl;
#pragma unroll
        for (int j = 0; j < 8; ++j) {
            const int row = rbase + j;
            if (row < M) {
                float* p = C + (size_t)row * ldc + col;
                if (accumulate) *p += acc[nt][j]; else *p = acc[nt][j];
            }
        }
    }
}

// ---------------- graph normalization / aggregation ------------------------

__global__ void deg_init(float* deg, int n) {
    int i = blockIdx.x * blockDim.x + threadIdx.x;
    if (i < n) deg[i] = 1.0f;                 // self loop
}

__global__ void deg_count(const int* __restrict__ dst, float* deg, int e) {
    int i = blockIdx.x * blockDim.x + threadIdx.x;
    if (i < e) atomicAdd(&deg[dst[i]], 1.0f);
}

__global__ void deg_rsqrt(float* deg, int n) {
    int i = blockIdx.x * blockDim.x + threadIdx.x;
    if (i < n) deg[i] = rsqrtf(fmaxf(deg[i], 1.0f));
}

// out[i][:] = Z[i][:] * dinv[i]^2   (self-loop term, also initializes out)
__global__ void agg_init(const float* __restrict__ Z, const float* __restrict__ dinv,
                         float* __restrict__ out, int n, int g4) {
    size_t tid = blockIdx.x * (size_t)blockDim.x + threadIdx.x;
    if (tid >= (size_t)n * g4) return;
    int i = (int)(tid / g4);
    float w = dinv[i] * dinv[i];
    float4 z = ((const float4*)Z)[tid];
    float4 o; o.x = z.x * w; o.y = z.y * w; o.z = z.z * w; o.w = z.w * w;
    ((float4*)out)[tid] = o;
}

// out[dst[e]][:] += Z[src[e]][:] * dinv[s]*dinv[d]
__global__ void agg_edges(const float* __restrict__ Z, const int* __restrict__ src,
                          const int* __restrict__ dst, const float* __restrict__ dinv,
                          float* __restrict__ out, int e, int g4) {
    size_t tid = blockIdx.x * (size_t)blockDim.x + threadIdx.x;
    if (tid >= (size_t)e * g4) return;
    int ed = (int)(tid / g4);
    int c  = (int)(tid % g4);
    int s  = src[ed], d = dst[ed];
    float w = dinv[s] * dinv[d];
    float4 z = ((const float4*)(Z + (size_t)s * (g4 * 4)))[c];
    float* o = out + (size_t)d * (g4 * 4) + c * 4;
    atomicAdd(o + 0, z.x * w);
    atomicAdd(o + 1, z.y * w);
    atomicAdd(o + 2, z.z * w);
    atomicAdd(o + 3, z.w * w);
}

__global__ void bias_act(float* __restrict__ out, const float* __restrict__ b,
                         int n, int g, int relu) {
    size_t tid = blockIdx.x * (size_t)blockDim.x + threadIdx.x;
    if (tid >= (size_t)n * g) return;
    int j = (int)(tid % g);
    float v = out[tid] + b[j];
    out[tid] = relu ? fmaxf(v, 0.0f) : v;
}

// ---------------- GRU elementwise ------------------------------------------

__global__ void gru_gates(float* __restrict__ tu, const float* __restrict__ bu,
                          float* __restrict__ tr, const float* __restrict__ br,
                          const float* __restrict__ h, float* __restrict__ rh,
                          int n, int hd) {
    size_t tid = blockIdx.x * (size_t)blockDim.x + threadIdx.x;
    if (tid >= (size_t)n * hd) return;
    int j = (int)(tid % hd);
    float u = 1.0f / (1.0f + __expf(-(tu[tid] + bu[j])));
    float r = 1.0f / (1.0f + __expf(-(tr[tid] + br[j])));
    tu[tid] = u;
    rh[tid] = r * h[tid];
}

__global__ void gru_hnew(const float* __restrict__ tc, const float* __restrict__ bc,
                         const float* __restrict__ u, float* __restrict__ h,
                         int n, int hd) {
    size_t tid = blockIdx.x * (size_t)blockDim.x + threadIdx.x;
    if (tid >= (size_t)n * hd) return;
    int j = (int)(tid % hd);
    float c  = tanhf(tc[tid] + bc[j]);
    float uu = u[tid];
    h[tid] = uu * h[tid] + (1.0f - uu) * c;
}

// out_t[i] = dot(h[i,:], Wo) + bo   (one wave32 per node)
__global__ void out_head(const float* __restrict__ h, const float* __restrict__ Wo,
                         const float* __restrict__ bo, float* __restrict__ out_t,
                         int n, int hd) {
    int wid  = (blockIdx.x * blockDim.x + threadIdx.x) >> 5;
    int lane = threadIdx.x & 31;
    if (wid >= n) return;
    float s = 0.0f;
    for (int k = lane; k < hd; k += 32) s += h[(size_t)wid * hd + k] * Wo[k];
#pragma unroll
    for (int off = 16; off; off >>= 1) s += __shfl_xor(s, off, 32);
    if (lane == 0) out_t[wid] = s + bo[0];
}

__global__ void zero_f(float* p, size_t n) {
    size_t tid = blockIdx.x * (size_t)blockDim.x + threadIdx.x;
    if (tid < n) p[tid] = 0.0f;
}

// ---------------------------------------------------------------------------

extern "C" void kernel_launch(void* const* d_in, const int* in_sizes, int n_in,
                              void* d_out, int out_size, void* d_ws, size_t ws_size,
                              hipStream_t stream) {
    constexpr int T = 8, N = 20000, F = 256, HD = 128, E = 320000, G = 3 * HD;

    const float* x   = (const float*)d_in[0];
    const int*   src = (const int*)d_in[1];
    const int*   dst = (const int*)d_in[2];
    const float* W1  = (const float*)d_in[3];
    const float* b1  = (const float*)d_in[4];
    const float* W2  = (const float*)d_in[5];
    const float* b2  = (const float*)d_in[6];
    const float* Wu  = (const float*)d_in[7];
    const float* bu  = (const float*)d_in[8];
    const float* Wr  = (const float*)d_in[9];
    const float* br  = (const float*)d_in[10];
    const float* Wc  = (const float*)d_in[11];
    const float* bc  = (const float*)d_in[12];
    const float* Wo  = (const float*)d_in[13];
    const float* bo  = (const float*)d_in[14];
    float* out = (float*)d_out;

    // Workspace layout (~113 MB)
    float* p    = (float*)d_ws;
    float* dinv = p; p += N;
    float* bufZ = p; p += (size_t)N * G;     // GEMM outputs (Z1 / Z2)
    float* bufH = p; p += (size_t)N * G;     // aggregated (H1 / URC)
    float* h    = p; p += (size_t)N * HD;
    float* tu   = p; p += (size_t)N * HD;
    float* tr   = p; p += (size_t)N * HD;
    float* tc   = p; p += (size_t)N * HD;
    float* rh   = p; p += (size_t)N * HD;

    const int BS = 256;
    auto blks = [](size_t n, int bs) { return (unsigned)((n + bs - 1) / bs); };

    zero_f<<<blks((size_t)N * HD, BS), BS, 0, stream>>>(h, (size_t)N * HD);

    dim3 gG((N + TM - 1) / TM, G / TN);    // 313 x 6
    dim3 gH((N + TM - 1) / TM, HD / TN);   // 313 x 2

    for (int t = 0; t < T; ++t) {
        const float* xt = x + (size_t)t * N * F;
        const int*   st = src + (size_t)t * E;
        const int*   dt = dst + (size_t)t * E;
        float* out_t = out + (size_t)t * N;

        // symmetric deg^-1/2 normalization (with self loops)
        deg_init <<<blks(N, BS), BS, 0, stream>>>(dinv, N);
        deg_count<<<blks(E, BS), BS, 0, stream>>>(dt, dinv, E);
        deg_rsqrt<<<blks(N, BS), BS, 0, stream>>>(dinv, N);

        // GCN layer 1: Z1 = x_t @ W1 ; H1 = relu(Aggregate(Z1) + b1)
        gemm_wmma_f16<<<gG, 128, 0, stream>>>(xt, F, W1, G, bufZ, G, N, F, G, 0);
        agg_init <<<blks((size_t)N * (G / 4), BS), BS, 0, stream>>>(bufZ, dinv, bufH, N, G / 4);
        agg_edges<<<blks((size_t)E * (G / 4), BS), BS, 0, stream>>>(bufZ, st, dt, dinv, bufH, E, G / 4);
        bias_act <<<blks((size_t)N * G, BS), BS, 0, stream>>>(bufH, b1, N, G, 1);

        // GCN layer 2: Z2 = H1 @ W2 ; URC = Aggregate(Z2) + b2
        gemm_wmma_f16<<<gG, 128, 0, stream>>>(bufH, G, W2, G, bufZ, G, N, G, G, 0);
        agg_init <<<blks((size_t)N * (G / 4), BS), BS, 0, stream>>>(bufZ, dinv, bufH, N, G / 4);
        agg_edges<<<blks((size_t)E * (G / 4), BS), BS, 0, stream>>>(bufZ, st, dt, dinv, bufH, E, G / 4);
        bias_act <<<blks((size_t)N * G, BS), BS, 0, stream>>>(bufH, b2, N, G, 0);

        // GRU gates: concat([gate_in, h]) @ W == gate_in@W_top + h@W_bot
        gemm_wmma_f16<<<gH, 128, 0, stream>>>(bufH,          G,  Wu,           HD, tu, HD, N, HD, HD, 0);
        gemm_wmma_f16<<<gH, 128, 0, stream>>>(h,             HD, Wu + HD * HD, HD, tu, HD, N, HD, HD, 1);
        gemm_wmma_f16<<<gH, 128, 0, stream>>>(bufH + HD,     G,  Wr,           HD, tr, HD, N, HD, HD, 0);
        gemm_wmma_f16<<<gH, 128, 0, stream>>>(h,             HD, Wr + HD * HD, HD, tr, HD, N, HD, HD, 1);
        gru_gates<<<blks((size_t)N * HD, BS), BS, 0, stream>>>(tu, bu, tr, br, h, rh, N, HD);
        gemm_wmma_f16<<<gH, 128, 0, stream>>>(bufH + 2 * HD, G,  Wc,           HD, tc, HD, N, HD, HD, 0);
        gemm_wmma_f16<<<gH, 128, 0, stream>>>(rh,            HD, Wc + HD * HD, HD, tc, HD, N, HD, HD, 1);
        gru_hnew<<<blks((size_t)N * HD, BS), BS, 0, stream>>>(tc, bc, tu, h, N, HD);

        out_head<<<blks((size_t)N * 32, BS), BS, 0, stream>>>(h, Wo, bo, out_t, N, HD);
    }

    // hT
    hipMemcpyAsync(out + (size_t)T * N, h, sizeof(float) * (size_t)N * HD,
                   hipMemcpyDeviceToDevice, stream);
}
